// AttnBlock_1975684956666
// MI455X (gfx1250) — matrix operand order repaired and checked by reading, not compiled, not run on hardware
//
#include <hip/hip_runtime.h>

#define B_   32
#define C_   256
#define N_   1024
#define G_   32
#define CPG  8          // channels per group
#define EPSV 1e-5f

typedef __attribute__((ext_vector_type(16))) _Float16 v16h;
typedef __attribute__((ext_vector_type(8)))  _Float16 v8h;
typedef __attribute__((ext_vector_type(8)))  float    v8f;

union V16U { v16h v; v8h h[2]; };

__device__ __forceinline__ v8f wmma16(v16h a, v16h b, v8f c) {
  return __builtin_amdgcn_wmma_f32_16x16x32_f16(false, a, false, b, (short)0, c, false, false);
}

// A fragment 16x32 (M=rows, K=cols) from row-major f16 src (leading dim ld halves).
// Lane layout: lane%16 = row; 16 contiguous halves at K = (hi?16:0)+0..15.
__device__ __forceinline__ v16h load_a(const _Float16* __restrict__ src, int ld,
                                       int r0, int k0, int lm, int hi8) {
  const _Float16* p = src + (size_t)(r0 + lm) * ld + (k0 + 2 * hi8);
  V16U u;
  u.h[0] = *reinterpret_cast<const v8h*>(p);
  u.h[1] = *reinterpret_cast<const v8h*>(p + 8);
  return u.v;
}

// B fragment 32x16 (K=rows, N=cols) where element (k,n) = src[n*ld + k] (i.e. src is [N, K]).
// Lane layout: lane%16 = column n; 16 contiguous halves at K = (hi?16:0)+0..15.
__device__ __forceinline__ v16h load_b(const _Float16* __restrict__ src, int ld,
                                       int n0, int k0, int lm, int hi16) {
  const _Float16* p = src + (size_t)(n0 + lm) * ld + (k0 + hi16);
  return *reinterpret_cast<const v16h*>(p);
}

// Async DMA: copy 16 bytes global -> LDS per lane (gfx1250 async-to-LDS path, ASYNCcnt).
__device__ __forceinline__ void async_copy16(unsigned lds_off, const _Float16* gptr) {
  asm volatile("global_load_async_to_lds_b128 %0, %1, off"
               :: "v"(lds_off), "v"(gptr) : "memory");
}
__device__ __forceinline__ void wait_async0() {
  asm volatile("s_wait_asynccnt 0x0" ::: "memory");
}

// ---------------- Kernel 0: fp32 -> f16 weight conversion ----------------
__global__ void wconv_kernel(const float* __restrict__ wq, const float* __restrict__ wk,
                             const float* __restrict__ wv, const float* __restrict__ wp,
                             _Float16* __restrict__ wh) {
  int i = blockIdx.x * 256 + threadIdx.x;          // 4*65536 total
  int m = i >> 16;
  int r = i & 65535;
  const float* src = (m == 0) ? wq : (m == 1) ? wk : (m == 2) ? wv : wp;
  wh[i] = (_Float16)src[r];
}

// ---------------- Kernel 1: GroupNorm -> hfT [B][N][C] f16 ----------------
__global__ void gn_kernel(const float* __restrict__ x, const float* __restrict__ gamma,
                          const float* __restrict__ beta, _Float16* __restrict__ hfT) {
  int b = blockIdx.x / G_;
  int g = blockIdx.x % G_;
  int tid = threadIdx.x;
  const float* xb = x + ((size_t)b * C_ + g * CPG) * N_;
  float s = 0.f, s2 = 0.f;
  for (int i = tid; i < CPG * N_; i += 256) {
    float v = xb[i];
    s += v; s2 += v * v;
  }
  __shared__ float r1[256], r2[256];
  r1[tid] = s; r2[tid] = s2;
  __syncthreads();
  for (int o = 128; o > 0; o >>= 1) {
    if (tid < o) { r1[tid] += r1[tid + o]; r2[tid] += r2[tid + o]; }
    __syncthreads();
  }
  const float invn = 1.f / (CPG * N_);
  float mean = r1[0] * invn;
  float var  = r2[0] * invn - mean * mean;
  float rstd = rsqrtf(var + EPSV);
  for (int i = tid; i < CPG * N_; i += 256) {
    int cl = i >> 10;           // N_ = 1024
    int n  = i & (N_ - 1);
    int c  = g * CPG + cl;
    float v = (xb[i] - mean) * rstd * gamma[c] + beta[c];
    hfT[((size_t)b * N_ + n) * C_ + c] = (_Float16)v;
  }
}

// ---------------- Kernel 2: QKV GEMMs (WMMA) ----------------
// q,k -> [B][N][C] f16 (q has softmax scale folded in); v -> [B][C][N] f16.
__global__ void qkv_kernel(const _Float16* __restrict__ hfT, const _Float16* __restrict__ wh,
                           const float* __restrict__ bq, const float* __restrict__ bk,
                           const float* __restrict__ bv,
                           _Float16* __restrict__ qT, _Float16* __restrict__ kT,
                           _Float16* __restrict__ vb) {
  int b  = blockIdx.x >> 6;
  int n0 = (blockIdx.x & 63) << 4;
  int wave = threadIdx.x >> 5;
  int lane = threadIdx.x & 31;
  int lm = lane & 15, hi = lane >> 4;
  const _Float16* hb = hfT + (size_t)b * N_ * C_;
  v16h Bf[8];
#pragma unroll
  for (int kk = 0; kk < 8; ++kk) Bf[kk] = load_b(hb, C_, n0, kk * 32, lm, hi * 16);
  const float scale = 0.0625f;  // 256^-0.5
  for (int t = wave; t < 48; t += 8) {
    int mat = t >> 4;
    int c0  = (t & 15) << 4;
    const _Float16* W = wh + (size_t)mat * C_ * C_;
    v16h Af[8];
#pragma unroll
    for (int kk = 0; kk < 8; ++kk) Af[kk] = load_a(W, C_, c0, kk * 32, lm, hi * 8);
    v8f acc = {};
#pragma unroll
    for (int kk = 0; kk < 8; ++kk) acc = wmma16(Af[kk], Bf[kk], acc);
    const float* bias = (mat == 0) ? bq : (mat == 1) ? bk : bv;
    int cbase = c0 + hi * 8;
    if (mat == 2) {
      _Float16* dst = vb + (size_t)b * C_ * N_;
#pragma unroll
      for (int v = 0; v < 8; ++v)
        dst[(size_t)(cbase + v) * N_ + n0 + lm] = (_Float16)(acc[v] + bias[cbase + v]);
    } else {
      _Float16* dst = ((mat == 0) ? qT : kT) + (size_t)b * N_ * C_;
      v8h o;
#pragma unroll
      for (int v = 0; v < 8; ++v) {
        float val = acc[v] + bias[cbase + v];
        if (mat == 0) val *= scale;
        o[v] = (_Float16)val;
      }
      *reinterpret_cast<v8h*>(dst + (size_t)(n0 + lm) * C_ + cbase) = o;
    }
  }
}

// ---------------- Kernel 3: fused flash attention ----------------
// 4 waves/block, each owns a 16-query tile. K/V chunks (32 keys) are staged into LDS
// via async-to-LDS DMA, double-buffered so the DMA of chunk mt+1 overlaps compute of mt.
// Sᵀ = K·qᵀ so softmax over keys is an in-register reduce + one xor-16 shuffle.
__global__ void __launch_bounds__(128)
attn_kernel(const _Float16* __restrict__ qT, const _Float16* __restrict__ kT,
            const _Float16* __restrict__ vb, _Float16* __restrict__ aoT) {
  __shared__ _Float16 Ks[2][32 * 256];   // [key][channel]   16 KB each
  __shared__ _Float16 Vs[2][256 * 32];   // [channel][key]   16 KB each

  int b = blockIdx.y;
  int wave = threadIdx.x >> 5;
  int lane = threadIdx.x & 31;
  int lm = lane & 15, hi = lane >> 4;
  int n0 = ((blockIdx.x << 2) + wave) << 4;
  const _Float16* qb = qT + (size_t)b * N_ * C_;
  const _Float16* kb = kT + (size_t)b * N_ * C_;
  const _Float16* vv = vb + (size_t)b * C_ * N_;

  // stage one 32-key chunk into LDS buffer `buf` via async DMA (all 128 threads)
  auto issue_chunk = [&](int mt, int buf) {
#pragma unroll
    for (int rep = 0; rep < 8; ++rep) {
      int idx = rep * 128 + threadIdx.x;          // 0..1023
      int km = idx >> 5, kc = (idx & 31) << 3;    // K: 32 rows x 256 halves
      async_copy16((unsigned)(size_t)&Ks[buf][km * 256 + kc],
                   kb + (size_t)(mt * 32 + km) * C_ + kc);
      int vc = idx >> 2, vm = (idx & 3) << 3;     // V: 256 rows x 32 halves
      async_copy16((unsigned)(size_t)&Vs[buf][vc * 32 + vm],
                   vv + (size_t)vc * N_ + mt * 32 + vm);
    }
  };

  v16h Qf[8];
#pragma unroll
  for (int kk = 0; kk < 8; ++kk) Qf[kk] = load_b(qb, C_, n0, kk * 32, lm, hi * 16);

  v8f O[16];
  v8f zero = {};
#pragma unroll
  for (int ct = 0; ct < 16; ++ct) O[ct] = zero;
  float mrun = -1e30f, lrun = 0.f;

  issue_chunk(0, 0);
  wait_async0();
  __syncthreads();

  for (int mt = 0; mt < 32; ++mt) {
    int buf = mt & 1;
    if (mt + 1 < 32) issue_chunk(mt + 1, buf ^ 1);   // DMA next chunk during compute
    const _Float16* ks = &Ks[buf][0];
    const _Float16* vs = &Vs[buf][0];

    v8f S0 = {}, S1 = {};
#pragma unroll
    for (int kk = 0; kk < 8; ++kk) {
      v16h A0 = load_a(ks, 256, 0, kk * 32, lm, hi * 8);
      S0 = wmma16(A0, Qf[kk], S0);
      v16h A1 = load_a(ks, 256, 16, kk * 32, lm, hi * 8);
      S1 = wmma16(A1, Qf[kk], S1);
    }
    // per-query (per-lane-column) running softmax over these 32 keys
    float tmax = -1e30f;
#pragma unroll
    for (int v = 0; v < 8; ++v) tmax = fmaxf(tmax, fmaxf(S0[v], S1[v]));
    tmax = fmaxf(tmax, __shfl_xor(tmax, 16, 32));
    float mnew  = fmaxf(mrun, tmax);
    float alpha = __expf(mrun - mnew);
    float p0[8], p1[8], rs = 0.f;
#pragma unroll
    for (int v = 0; v < 8; ++v) {
      p0[v] = __expf(S0[v] - mnew);
      p1[v] = __expf(S1[v] - mnew);
      rs += p0[v] + p1[v];
    }
    rs += __shfl_xor(rs, 16, 32);
    lrun = lrun * alpha + rs;
    mrun = mnew;
#pragma unroll
    for (int ct = 0; ct < 16; ++ct)
#pragma unroll
      for (int v = 0; v < 8; ++v) O[ct][v] *= alpha;
    // Pack Pᵀ (keys x queries) into a B-fragment: lanes<16 take keys 0..15 (own St0 rows 0..7
    // + partner's rows 8..15), lanes>=16 take keys 16..31 (partner St1 rows 0..7 + own 8..15).
    V16U P;
#pragma unroll
    for (int v = 0; v < 8; ++v) {
      float c0f = __shfl_xor(p0[v], 16, 32);
      float c1f = __shfl_xor(p1[v], 16, 32);
      P.h[0][v] = (_Float16)(hi ? c1f : p0[v]);
      P.h[1][v] = (_Float16)(hi ? p1[v] : c0f);
    }
#pragma unroll
    for (int ct = 0; ct < 16; ++ct) {
      v16h Vf = load_a(vs, 32, ct * 16, 0, lm, hi * 8);  // A = V chunk [C][32 keys]
      O[ct] = wmma16(Vf, P.v, O[ct]);
    }
    wait_async0();      // next chunk landed (overlapped with the compute above)
    __syncthreads();    // all waves done with `buf` before it is overwritten
  }

  float inv = 1.f / lrun;
  _Float16* dst = aoT + (size_t)b * N_ * C_;
#pragma unroll
  for (int ct = 0; ct < 16; ++ct) {
    v8h o;
#pragma unroll
    for (int v = 0; v < 8; ++v) o[v] = (_Float16)(O[ct][v] * inv);
    *reinterpret_cast<v8h*>(dst + (size_t)(n0 + lm) * C_ + ct * 16 + hi * 8) = o;
  }
}

// ---------------- Kernel 4: proj GEMM + bias + residual ----------------
__global__ void proj_kernel(const _Float16* __restrict__ aoT, const _Float16* __restrict__ whp,
                            const float* __restrict__ bp, const float* __restrict__ x,
                            float* __restrict__ out) {
  int wave = threadIdx.x >> 5;
  int lane = threadIdx.x & 31;
  int lm = lane & 15, hi = lane >> 4;
  int t  = blockIdx.x * 8 + wave;  // B*16*64 = 32768 tiles
  int b  = t >> 10;
  int ct = (t >> 6) & 15;
  int nt = t & 63;
  int c0 = ct << 4, n0 = nt << 4;
  const _Float16* ab = aoT + (size_t)b * N_ * C_;
  v16h Af[8], Bf[8];
#pragma unroll
  for (int kk = 0; kk < 8; ++kk) {
    Af[kk] = load_a(whp, C_, c0, kk * 32, lm, hi * 8);
    Bf[kk] = load_b(ab, C_, n0, kk * 32, lm, hi * 16);
  }
  v8f acc = {};
#pragma unroll
  for (int kk = 0; kk < 8; ++kk) acc = wmma16(Af[kk], Bf[kk], acc);
  int cbase = c0 + hi * 8;
  size_t base = ((size_t)b * C_ + cbase) * N_ + n0 + lm;
#pragma unroll
  for (int v = 0; v < 8; ++v) {
    size_t idx = base + (size_t)v * N_;
    out[idx] = x[idx] + acc[v] + bp[cbase + v];
  }
}

extern "C" void kernel_launch(void* const* d_in, const int* in_sizes, int n_in,
                              void* d_out, int out_size, void* d_ws, size_t ws_size,
                              hipStream_t stream) {
  const float* x  = (const float*)d_in[0];
  const float* gs = (const float*)d_in[1];
  const float* gb = (const float*)d_in[2];
  const float* wq = (const float*)d_in[3];
  const float* bq = (const float*)d_in[4];
  const float* wk = (const float*)d_in[5];
  const float* bk = (const float*)d_in[6];
  const float* wv = (const float*)d_in[7];
  const float* bv = (const float*)d_in[8];
  const float* wp = (const float*)d_in[9];
  const float* bp = (const float*)d_in[10];
  float* out = (float*)d_out;

  char* ws = (char*)d_ws;
  const size_t halfmat = (size_t)B_ * N_ * C_ * sizeof(_Float16);  // 16 MiB each
  _Float16* hfT = (_Float16*)(ws);
  _Float16* qT  = (_Float16*)(ws + 1 * halfmat);
  _Float16* kT  = (_Float16*)(ws + 2 * halfmat);
  _Float16* vb  = (_Float16*)(ws + 3 * halfmat);
  _Float16* aoT = (_Float16*)(ws + 4 * halfmat);
  _Float16* wh  = (_Float16*)(ws + 5 * halfmat);  // 4 x [256x256] f16

  wconv_kernel<<<1024, 256, 0, stream>>>(wq, wk, wv, wp, wh);
  gn_kernel<<<B_ * G_, 256, 0, stream>>>(x, gs, gb, hfT);
  qkv_kernel<<<B_ * 64, 256, 0, stream>>>(hfT, wh, bq, bk, bv, qT, kT, vb);
  attn_kernel<<<dim3(16, B_), 128, 0, stream>>>(qT, kT, vb, aoT);
  proj_kernel<<<4096, 256, 0, stream>>>(aoT, wh + 3 * (size_t)C_ * C_, bp, x, out);
}
